// RNES_24103356465418
// MI455X (gfx1250) — matrix operand order repaired and checked by reading, not compiled, not run on hardware
//
#include <hip/hip_runtime.h>
#include <hip/hip_bf16.h>
#include <math.h>

// Problem constants (from reference)
#define S_N   512
#define WL_N  100
#define E_N   200
#define OC_N  50
#define H_N   400
#define G3H   1200     // 3*H
#define TPAD  128      // padded time extent of gathered embeddings (halo + spill safety)
#define PADW  7        // max conv half-width
#define KPAD  416      // K for gi GEMM, padded to multiple of 32
#define NT_IH 75       // 1200/16 N-tiles for gi GEMM
#define KT_IH 13       // ceil(400/32) K-tiles for gi GEMM

typedef __bf16 bf16_t;
typedef __attribute__((ext_vector_type(8)))  __bf16 v8bf;
typedef __attribute__((ext_vector_type(16))) __bf16 v16bf;
typedef __attribute__((ext_vector_type(8)))  float  v8f;

static __device__ __forceinline__ bf16_t f2bf(float f) {
  unsigned u = __builtin_bit_cast(unsigned, f);
  unsigned r = u + 0x7fffu + ((u >> 16) & 1u);  // round-to-nearest-even
  unsigned short h = (unsigned short)(r >> 16);
  return __builtin_bit_cast(bf16_t, h);
}
static __device__ __forceinline__ float sigm(float x) { return 1.0f / (1.0f + expf(-x)); }
static __device__ __forceinline__ int conv_ktiles(int c) { return ((2 * c + 1) * E_N + 31) >> 5; }
static __device__ __forceinline__ int conv_bbase(int c) {
  int b = 0;
  for (int i = 0; i < c; ++i) b += conv_ktiles(i) * 4 * 32 * 16;
  return b;
}
static __device__ __forceinline__ v16bf cat8(v8bf a, v8bf b) {
  return __builtin_shufflevector(a, b, 0,1,2,3,4,5,6,7,8,9,10,11,12,13,14,15);
}

// ---------------------------------------------------------------------------
// 1) lengths[s] = sum(x[s,:] > 0)
__global__ void rnes_lengths_kernel(const int* __restrict__ x, int* __restrict__ lengths) {
  __shared__ int cnt;
  int s = blockIdx.x, t = threadIdx.x;
  if (t == 0) cnt = 0;
  __syncthreads();
  if (t < WL_N && x[s * WL_N + t] > 0) atomicAdd(&cnt, 1);
  __syncthreads();
  if (t == 0) lengths[s] = cnt;
}

// ---------------------------------------------------------------------------
// 2) gather bf16 embeddings with zero halo: wf[s][tp][e], tp in [0,TPAD)
//    tp in [PADW, PADW+WL) -> emb[x[s][tp-PADW]] (token 0 gathers emb[0], as in ref)
__global__ void rnes_gather_kernel(const int* __restrict__ x, const float* __restrict__ emb,
                                   bf16_t* __restrict__ wf) {
  int tp = blockIdx.x, s = blockIdx.y, e = threadIdx.x;
  if (e >= E_N) return;
  float v = 0.0f;
  int t = tp - PADW;
  if (t >= 0 && t < WL_N) {
    int tok = x[s * WL_N + t];
    v = emb[(size_t)tok * E_N + e];
  }
  wf[((size_t)s * TPAD + tp) * E_N + e] = f2bf(v);
}

// ---------------------------------------------------------------------------
// 3) pack conv weights into WMMA B-operand lane layout, zero-filled tails.
//    B[r][n] with r = k*E + e ; lane<16: K-chunk [kt*32..+15], lane>=16: [+16..+31]
__global__ void rnes_pack_conv_kernel(const float* w0, const float* w1, const float* w2,
                                      const float* w3, const float* w4, const float* w5,
                                      const float* w6, const float* w7, bf16_t* __restrict__ Bp) {
  const float* ws[8] = {w0, w1, w2, w3, w4, w5, w6, w7};
  int slot = blockIdx.x;
  int c = 0;
  while (slot >= conv_ktiles(c)) { slot -= conv_ktiles(c); ++c; }
  int kt = slot;
  int ntile = threadIdx.x >> 5, lane = threadIdx.x & 31;
  int Kw = 2 * c + 1, Rc = Kw * E_N;
  int n = ntile * 16 + (lane & 15);
  int kbase = kt * 32 + ((lane & 16) ? 16 : 0);
  bf16_t* dst = Bp + conv_bbase(c) + ((size_t)(kt * 4 + ntile) * 32 + lane) * 16;
  const float* w = ws[c];
  for (int j = 0; j < 16; ++j) {
    int r = kbase + j;
    float v = 0.0f;
    if (n < OC_N && r < Rc) {
      int k = r / E_N, e = r % E_N;
      v = w[(size_t)n * E_N * Kw + (size_t)e * Kw + k];
    }
    dst[j] = f2bf(v);
  }
}

// ---------------------------------------------------------------------------
// 4) pack w_ih (both dirs) for gi GEMM: B[k][n] = w_ih[n][k], K padded to 416
__global__ void rnes_pack_ih_kernel(const float* __restrict__ w_ih_f,
                                    const float* __restrict__ w_ih_b, bf16_t* __restrict__ Bp) {
  int slot = blockIdx.x, dir = blockIdx.y, lane = threadIdx.x;
  const float* w = dir ? w_ih_b : w_ih_f;
  int ntile = slot / KT_IH;
  int kt = slot % KT_IH;
  int n = ntile * 16 + (lane & 15);
  int kbase = kt * 32 + ((lane & 16) ? 16 : 0);
  bf16_t* dst = Bp + ((size_t)dir * NT_IH * KT_IH + (size_t)(ntile * KT_IH + kt)) * 32 * 16 +
                (size_t)lane * 16;
  for (int j = 0; j < 16; ++j) {
    int k = kbase + j;
    dst[j] = f2bf((k < H_N) ? w[(size_t)n * H_N + k] : 0.0f);
  }
}

// ---------------------------------------------------------------------------
// 5) conv bank as WMMA GEMM with fused masked ragged average-pool.
//    grid=(8 convs, 512 sents), 4 waves: wave = N-tile (16 channels).
//    im2col row for time t is a contiguous slice of wf (time-contiguous layout),
//    so A loads are direct, aligned b128 loads; B tails are zero-packed.
__global__ void rnes_conv_pool_kernel(const bf16_t* __restrict__ wf, const bf16_t* __restrict__ Bp,
                                      const float* __restrict__ conv_b,
                                      const int* __restrict__ lengths, float* __restrict__ pooled) {
  int c = blockIdx.x, s = blockIdx.y;
  int wave = threadIdx.x >> 5, lane = threadIdx.x & 31;
  int m = lane & 15, hi = lane >> 4;
  int len = lengths[s];
  int ktn = conv_ktiles(c);
  const bf16_t* Bc = Bp + conv_bbase(c);
  float colsum = 0.0f;
  for (int tile = 0; tile < 7; ++tile) {
    int t0 = tile * 16;
    const bf16_t* Arow = wf + ((size_t)s * TPAD + (size_t)(t0 + m + PADW - c)) * E_N;
    v8f acc = {};
    for (int kt = 0; kt < ktn; ++kt) {
      int ko = kt * 32;
      v8bf a0 = *(const v8bf*)(Arow + ko + hi * 16);
      v8bf a1 = *(const v8bf*)(Arow + ko + hi * 16 + 8);
      v16bf a = cat8(a0, a1);
      const bf16_t* bp = Bc + ((size_t)(kt * 4 + wave) * 32 + lane) * 16;
      __builtin_prefetch(bp + 4 * 32 * 16, 0, 0);  // next K-tile of B
      v8bf b0 = *(const v8bf*)bp;
      v8bf b1 = *(const v8bf*)(bp + 8);
      v16bf b = cat8(b0, b1);
      acc = __builtin_amdgcn_wmma_f32_16x16x32_bf16(false, a, false, b, (short)0, acc,
                                                    false, false);
    }
    // C/D layout: lane -> N = lane&15 ; vgpr j -> M = j + (hi?8:0)
    for (int j = 0; j < 8; ++j) {
      int t = t0 + j + hi * 8;
      if (t < len) colsum += acc[j];
    }
  }
  colsum += __shfl_xor(colsum, 16, 32);  // fold the two M-halves per N column
  int ch = wave * 16 + m;
  if (hi == 0 && ch < OC_N) {
    pooled[(size_t)s * H_N + c * OC_N + ch] = colsum / (float)len + conv_b[c * OC_N + ch];
  }
}

// ---------------------------------------------------------------------------
// 6) pooled f32 -> bf16, K padded to 416 with zeros
__global__ void rnes_pooled_cvt_kernel(const float* __restrict__ pooled, bf16_t* __restrict__ pbf) {
  int s = blockIdx.x, i = threadIdx.x;
  if (i >= KPAD) return;
  float v = (i < H_N) ? pooled[(size_t)s * H_N + i] : 0.0f;
  pbf[(size_t)s * KPAD + i] = f2bf(v);
}

// ---------------------------------------------------------------------------
// 7) gi[dir] = pooled @ w_ih^T + b_ih  : [512 x 400] x [400 x 1200], WMMA bf16
__global__ void rnes_gi_gemm_kernel(const bf16_t* __restrict__ pbf, const bf16_t* __restrict__ Bp,
                                    const float* __restrict__ b_ih_f,
                                    const float* __restrict__ b_ih_b, float* __restrict__ gi) {
  int wave = threadIdx.x >> 5, lane = threadIdx.x & 31;
  int ntile = blockIdx.x * 4 + wave;
  if (ntile >= NT_IH) return;  // wave-uniform exit
  int mtile = blockIdx.y, dir = blockIdx.z;
  int m = lane & 15, hi = lane >> 4;
  const bf16_t* Arow = pbf + (size_t)(mtile * 16 + m) * KPAD;
  const bf16_t* Bd = Bp + (size_t)dir * NT_IH * KT_IH * 32 * 16;
  v8f acc = {};
  for (int kt = 0; kt < KT_IH; ++kt) {
    int ko = kt * 32;
    v8bf a0 = *(const v8bf*)(Arow + ko + hi * 16);
    v8bf a1 = *(const v8bf*)(Arow + ko + hi * 16 + 8);
    v16bf a = cat8(a0, a1);
    const bf16_t* bp = Bd + ((size_t)(ntile * KT_IH + kt) * 32 + lane) * 16;
    v8bf b0 = *(const v8bf*)bp;
    v8bf b1 = *(const v8bf*)(bp + 8);
    v16bf b = cat8(b0, b1);
    acc = __builtin_amdgcn_wmma_f32_16x16x32_bf16(false, a, false, b, (short)0, acc, false, false);
  }
  const float* bih = dir ? b_ih_b : b_ih_f;
  int n = ntile * 16 + m;
  float bn = bih[n];
  for (int j = 0; j < 8; ++j) {
    int srow = mtile * 16 + j + hi * 8;
    gi[((size_t)dir * S_N + srow) * G3H + n] = acc[j] + bn;
  }
}

// ---------------------------------------------------------------------------
// 8) sequential bidirectional GRU scan. One block per direction.
__global__ void rnes_gru_scan_kernel(const float* __restrict__ gi, const float* __restrict__ whhf,
                                     const float* __restrict__ whhb, const float* __restrict__ bhhf,
                                     const float* __restrict__ bhhb, float* __restrict__ enc) {
  int dir = blockIdx.x;
  const float* whh = dir ? whhb : whhf;
  const float* bhh = dir ? bhhb : bhhf;
  const float* giD = gi + (size_t)dir * S_N * G3H;
  __shared__ float h[H_N];
  __shared__ float gh[G3H];
  int tid = threadIdx.x;
  for (int j = tid; j < H_N; j += blockDim.x) h[j] = 0.0f;
  __syncthreads();
  for (int step = 0; step < S_N; ++step) {
    int s = dir ? (S_N - 1 - step) : step;
    for (int j = tid; j < G3H; j += blockDim.x) {
      const float* row = whh + (size_t)j * H_N;
      float acc = bhh[j];
      for (int k = 0; k < H_N; ++k) acc += row[k] * h[k];
      gh[j] = acc;
    }
    __syncthreads();
    float hn = 0.0f;
    if (tid < H_N) {
      const float* g = giD + (size_t)s * G3H;
      float r = sigm(g[tid] + gh[tid]);
      float z = sigm(g[H_N + tid] + gh[H_N + tid]);
      float n = tanhf(g[2 * H_N + tid] + r * gh[2 * H_N + tid]);
      hn = (1.0f - z) * n + z * h[tid];
      enc[(size_t)s * (2 * H_N) + dir * H_N + tid] = hn;
    }
    __syncthreads();
    if (tid < H_N) h[tid] = hn;
    __syncthreads();
  }
}

// ---------------------------------------------------------------------------
// 9) doc = tanh(W_doc @ mean(enc, axis=0) + b_doc)
__global__ void rnes_doc_kernel(const float* __restrict__ enc, const float* __restrict__ W_doc,
                                const float* __restrict__ b_doc, float* __restrict__ docv) {
  __shared__ float mean[2 * H_N];
  int tid = threadIdx.x;
  for (int f = tid; f < 2 * H_N; f += blockDim.x) {
    float acc = 0.0f;
    for (int s = 0; s < S_N; ++s) acc += enc[(size_t)s * (2 * H_N) + f];
    mean[f] = acc * (1.0f / (float)S_N);
  }
  __syncthreads();
  for (int j = tid; j < H_N; j += blockDim.x) {
    const float* row = W_doc + (size_t)j * (2 * H_N);
    float acc = b_doc[j];
    for (int k = 0; k < 2 * H_N; ++k) acc += row[k] * mean[k];
    docv[j] = tanhf(acc);
  }
}

// ---------------------------------------------------------------------------
// 10) sequential greedy decoder; logp -> out[0] (float), sels -> out[1..] (int32 bits)
__global__ void rnes_decoder_kernel(const float* __restrict__ enc, const float* __restrict__ docv,
                                    const float* __restrict__ W_d1, const float* __restrict__ b_d1,
                                    const float* __restrict__ W_d2, const float* __restrict__ b_d2,
                                    const float* __restrict__ W_r, const float* __restrict__ b_r,
                                    const int* __restrict__ nos_p, float* __restrict__ out) {
  __shared__ float feat[1600];  // [enc_i(800) | g(400) | doc(400)]
  __shared__ float hid[100];
  __shared__ int selS;
  int tid = threadIdx.x;
  int nos = nos_p[0];
  for (int j = tid; j < H_N; j += blockDim.x) {
    feat[800 + j] = 0.0f;        // g = 0
    feat[1200 + j] = docv[j];    // doc
  }
  float logp = 0.0f;
  int count = 0;
  for (int s = 0; s < S_N; ++s) {
    for (int f = tid; f < 800; f += blockDim.x) feat[f] = enc[(size_t)s * 800 + f];
    __syncthreads();
    for (int j = tid; j < 100; j += blockDim.x) {
      const float* row = W_d1 + (size_t)j * 1600;
      float acc = b_d1[j];
      for (int k = 0; k < 1600; ++k) acc += row[k] * feat[k];
      hid[j] = tanhf(acc);
    }
    __syncthreads();
    if (tid == 0) {
      float acc = b_d2[0];
      for (int k = 0; k < 100; ++k) acc += W_d2[k] * hid[k];
      float p = sigm(acc);
      bool allowed = (nos <= 0) || (count < nos);
      int sel = (allowed && p > 0.5f) ? 1 : 0;
      count += sel;
      float term = sel ? p : (1.0f - p);
      logp += logf(term * 0.99999f + 5e-6f);
      selS = sel;
      ((int*)out)[1 + s] = sel;  // int32 output written as raw bits
    }
    __syncthreads();
    if (selS) {
      for (int j = tid; j < H_N; j += blockDim.x) {
        const float* row = W_r + (size_t)j * 800;
        float acc = b_r[j];
        for (int k = 0; k < 800; ++k) acc += row[k] * feat[k];  // feat[0:800] = enc_i
        feat[800 + j] += tanhf(acc);  // g += tanh(W_r @ enc_i + b_r)
      }
    }
    __syncthreads();
  }
  if (tid == 0) out[0] = logp;
}

// ---------------------------------------------------------------------------
extern "C" void kernel_launch(void* const* d_in, const int* in_sizes, int n_in,
                              void* d_out, int out_size, void* d_ws, size_t ws_size,
                              hipStream_t stream) {
  (void)in_sizes; (void)n_in; (void)out_size; (void)ws_size;
  const int*   x      = (const int*)d_in[0];
  const int*   nos    = (const int*)d_in[1];
  const float* emb    = (const float*)d_in[2];
  const float* cw[8];
  for (int i = 0; i < 8; ++i) cw[i] = (const float*)d_in[3 + i];
  const float* conv_b = (const float*)d_in[11];
  const float* w_ih_f = (const float*)d_in[12];
  const float* w_hh_f = (const float*)d_in[13];
  const float* b_ih_f = (const float*)d_in[14];
  const float* b_hh_f = (const float*)d_in[15];
  const float* w_ih_b = (const float*)d_in[16];
  const float* w_hh_b = (const float*)d_in[17];
  const float* b_ih_b = (const float*)d_in[18];
  const float* b_hh_b = (const float*)d_in[19];
  const float* W_doc  = (const float*)d_in[20];
  const float* b_doc  = (const float*)d_in[21];
  const float* W_d1   = (const float*)d_in[22];
  const float* b_d1   = (const float*)d_in[23];
  const float* W_d2   = (const float*)d_in[24];
  const float* b_d2   = (const float*)d_in[25];
  const float* W_r    = (const float*)d_in[26];
  const float* b_r    = (const float*)d_in[27];

  // workspace carve-up (256B aligned)
  int conv_kt_total = 0;
  for (int c = 0; c < 8; ++c) conv_kt_total += ((2 * c + 1) * E_N + 31) >> 5;  // = 404
  size_t off = 0;
  auto carve = [&](size_t bytes) {
    size_t o = off;
    off = (off + bytes + 255) & ~(size_t)255;
    return o;
  };
  char* base = (char*)d_ws;
  bf16_t* wf     = (bf16_t*)(base + carve((size_t)S_N * TPAD * E_N * sizeof(bf16_t)));
  bf16_t* BpC    = (bf16_t*)(base + carve((size_t)conv_kt_total * 4 * 32 * 16 * sizeof(bf16_t)));
  bf16_t* BpI    = (bf16_t*)(base + carve((size_t)2 * NT_IH * KT_IH * 32 * 16 * sizeof(bf16_t)));
  int*    lens   = (int*)   (base + carve((size_t)S_N * sizeof(int)));
  float*  pooled = (float*) (base + carve((size_t)S_N * H_N * sizeof(float)));
  bf16_t* pbf    = (bf16_t*)(base + carve((size_t)S_N * KPAD * sizeof(bf16_t)));
  float*  gi     = (float*) (base + carve((size_t)2 * S_N * G3H * sizeof(float)));
  float*  enc    = (float*) (base + carve((size_t)S_N * 2 * H_N * sizeof(float)));
  float*  docv   = (float*) (base + carve((size_t)H_N * sizeof(float)));

  rnes_lengths_kernel<<<S_N, 128, 0, stream>>>(x, lens);
  rnes_gather_kernel<<<dim3(TPAD, S_N), 256, 0, stream>>>(x, emb, wf);
  rnes_pack_conv_kernel<<<conv_kt_total, 128, 0, stream>>>(cw[0], cw[1], cw[2], cw[3], cw[4],
                                                           cw[5], cw[6], cw[7], BpC);
  rnes_pack_ih_kernel<<<dim3(NT_IH * KT_IH, 2), 32, 0, stream>>>(w_ih_f, w_ih_b, BpI);
  rnes_conv_pool_kernel<<<dim3(8, S_N), 128, 0, stream>>>(wf, BpC, conv_b, lens, pooled);
  rnes_pooled_cvt_kernel<<<S_N, 448, 0, stream>>>(pooled, pbf);
  rnes_gi_gemm_kernel<<<dim3(19, S_N / 16, 2), 128, 0, stream>>>(pbf, BpI, b_ih_f, b_ih_b, gi);
  rnes_gru_scan_kernel<<<2, 1024, 0, stream>>>(gi, w_hh_f, w_hh_b, b_hh_f, b_hh_b, enc);
  rnes_doc_kernel<<<1, 1024, 0, stream>>>(enc, W_doc, b_doc, docv);
  rnes_decoder_kernel<<<1, 1024, 0, stream>>>(enc, docv, W_d1, b_d1, W_d2, b_d2, W_r, b_r, nos,
                                              (float*)d_out);
}
